// SAGE_86766929314085
// MI455X (gfx1250) — matrix operand order repaired and checked by reading, not compiled, not run on hardware
//
#include <hip/hip_runtime.h>
#include <hip/hip_bf16.h>

#define N0v   614400
#define N1v   40960
#define Bv    4096
#define F0v   15
#define F1v   10
#define INF   128
#define NH    256
#define NG    1024   // 4*NH
#define NC    32
#define BN_EPS 1e-5f

typedef __attribute__((ext_vector_type(16))) __bf16 bf16x16;
typedef __attribute__((ext_vector_type(8)))  __bf16 bf16x8;
typedef __attribute__((ext_vector_type(4)))  __bf16 bf16x4;
typedef __attribute__((ext_vector_type(8)))  float  f32x8;

union BF16Frag { bf16x16 v; bf16x8 h[2]; };

__device__ __forceinline__ f32x8 wmma_bf16(bf16x16 a, bf16x16 b, f32x8 c) {
    return __builtin_amdgcn_wmma_f32_16x16x32_bf16(false, a, false, b, (short)0, c, false, false);
}

__device__ __forceinline__ float sigm(float x) { return 1.f / (1.f + expf(-x)); }

// ---- optional CDNA5 async global->LDS copy path (graceful fallback) ----
#if __has_builtin(__builtin_amdgcn_global_load_async_to_lds_b128)
#define HAVE_ASYNC_LDS 1
typedef int i32x4_vs __attribute__((vector_size(4 * sizeof(int))));
typedef __attribute__((address_space(1))) i32x4_vs as1_i32x4;
typedef __attribute__((address_space(3))) i32x4_vs as3_i32x4;
__device__ __forceinline__ void async_copy_b128(void* lds, const void* gptr) {
    __builtin_amdgcn_global_load_async_to_lds_b128(
        (as1_i32x4*)(gptr),
        (as3_i32x4*)(lds),
        0, 0);
}
__device__ __forceinline__ void wait_async0() {
#if __has_builtin(__builtin_amdgcn_s_wait_asynccnt)
    __builtin_amdgcn_s_wait_asynccnt(0);
#else
    asm volatile("s_wait_asynccnt 0" ::: "memory");
#endif
}
#endif

// ---------------- utility kernels ----------------

__global__ void k_cvt_bf16(const float* __restrict__ s, __bf16* __restrict__ d, int n) {
    int i = blockIdx.x * blockDim.x + threadIdx.x;
    if (i < n) d[i] = (__bf16)s[i];
}

__global__ void k_zero16(uint4* __restrict__ p, size_t n16) {
    size_t i = (size_t)blockIdx.x * blockDim.x + threadIdx.x;
    size_t stride = (size_t)gridDim.x * blockDim.x;
    uint4 z; z.x = z.y = z.z = z.w = 0u;
    for (; i < n16; i += stride) p[i] = z;
}

// ---------------- stage 1: hp = relu(log1p(x) @ W_pool^T + b_pool), bf16 out ----------------
// block: 256 thr (8 waves), computes 16 rows x 128 cols. K = 128.
// Output strip is staged back through LDS so hp stores are coalesced b128.
__global__ __launch_bounds__(256) void k_pool(const float* __restrict__ x,
                                              const __bf16* __restrict__ Wp,
                                              const float* __restrict__ bpool,
                                              __bf16* __restrict__ hp) {
    __shared__ __align__(16) __bf16 As[16][136]; // +8 bf16 pad: keeps 16B align, breaks bank collide
    const int row0 = blockIdx.x * 16;
    const int tid  = threadIdx.x;
    { // cooperative load + log1p + cvt
        int r = tid >> 4;            // 0..15
        int c = (tid & 15) * 8;      // 0..120
        const float4* p = (const float4*)&x[(size_t)(row0 + r) * INF + c];
        float4 a = p[0], b = p[1];
        __bf16* d = &As[r][c];
        d[0] = (__bf16)logf(a.x + 1.f); d[1] = (__bf16)logf(a.y + 1.f);
        d[2] = (__bf16)logf(a.z + 1.f); d[3] = (__bf16)logf(a.w + 1.f);
        d[4] = (__bf16)logf(b.x + 1.f); d[5] = (__bf16)logf(b.y + 1.f);
        d[6] = (__bf16)logf(b.z + 1.f); d[7] = (__bf16)logf(b.w + 1.f);
    }
    __syncthreads();
    const int wave = tid >> 5, lane = tid & 31;
    const int n0 = wave * 16;
    const int lr = lane & 15, lh = lane >> 4;
    f32x8 acc = {};
#pragma unroll
    for (int kk = 0; kk < 4; ++kk) {
        const int k0 = kk * 32;
        BF16Frag a, b;
        a.h[0] = *(const bf16x8*)&As[lr][k0 + lh * 8];
        a.h[1] = *(const bf16x8*)&As[lr][k0 + lh * 8 + 16];
        const __bf16* bp = &Wp[(size_t)(n0 + lr) * INF + k0 + lh * 8];
        b.h[0] = *(const bf16x8*)bp;
        b.h[1] = *(const bf16x8*)(bp + 16);
        acc = wmma_bf16(a.v, b.v, acc);
    }
    __syncthreads();   // all A-fragment reads done; safe to reuse As as C-staging
    {
        const int col = n0 + lr;
        const float bias = bpool[col];
#pragma unroll
        for (int v = 0; v < 8; ++v) {
            int row = v + lh * 8;
            float val = acc[v] + bias;
            As[row][col] = (__bf16)(val > 0.f ? val : 0.f);
        }
    }
    __syncthreads();
    { // coalesced b128 row-major store
        int r = tid >> 4;
        int c = (tid & 15) * 8;
        *(bf16x8*)&hp[(size_t)(row0 + r) * INF + c] = *(const bf16x8*)&As[r][c];
    }
}

// ---------------- stage 2: neighbor max-pool gather (prefetch 2 rows ahead) --------------
__global__ __launch_bounds__(128) void k_max(const __bf16* __restrict__ hp,
                                             const int* __restrict__ nbr0,
                                             __bf16* __restrict__ hneigh) {
    __shared__ int sidx[F0v];
    const int row = blockIdx.x, j = threadIdx.x;
    if (j < F0v) sidx[j] = nbr0[row * F0v + j];
    __syncthreads();
    __builtin_prefetch(&hp[(size_t)sidx[0] * INF + j], 0, 3);
    __builtin_prefetch(&hp[(size_t)sidx[1] * INF + j], 0, 3);
    float m = -3.4e38f;
    for (int f = 0; f < F0v; ++f) {
        if (f + 2 < F0v) __builtin_prefetch(&hp[(size_t)sidx[f + 2] * INF + j], 0, 3);
        float v = (float)hp[(size_t)sidx[f] * INF + j];
        m = fmaxf(m, v);
    }
    hneigh[(size_t)row * INF + j] = (__bf16)m;
}

// ---------------- stage 3: h0 = log1p(x[:N1]) @ Ws^T + hneigh @ Wn^T + b0 (fp32 out) ----------
// block: 512 thr (16 waves), 16 rows x 256 cols, one 16x16 tile per wave, K=128 per source.
__global__ __launch_bounds__(512) void k_layer0(const float* __restrict__ x,
                                                const __bf16* __restrict__ hneigh,
                                                const __bf16* __restrict__ Ws,
                                                const __bf16* __restrict__ Wn,
                                                const float* __restrict__ b0,
                                                float* __restrict__ h0) {
    __shared__ __align__(16) __bf16 Aself[16][136];
    __shared__ __align__(16) __bf16 Anei[16][136];
    const int row0 = blockIdx.x * 16;
    const int tid  = threadIdx.x;
    {
        int r = tid >> 5;            // 0..15
        int c = (tid & 31) * 4;      // 0..124
        float4 a = *(const float4*)&x[(size_t)(row0 + r) * INF + c];
        Aself[r][c + 0] = (__bf16)logf(a.x + 1.f);
        Aself[r][c + 1] = (__bf16)logf(a.y + 1.f);
        Aself[r][c + 2] = (__bf16)logf(a.z + 1.f);
        Aself[r][c + 3] = (__bf16)logf(a.w + 1.f);
        *(bf16x4*)&Anei[r][c] = *(const bf16x4*)&hneigh[(size_t)(row0 + r) * INF + c];
    }
    __syncthreads();
    const int wave = tid >> 5, lane = tid & 31;
    const int n0 = wave * 16;
    const int lr = lane & 15, lh = lane >> 4;
    f32x8 acc = {};
#pragma unroll
    for (int kk = 0; kk < 4; ++kk) {
        const int k0 = kk * 32;
        BF16Frag a, b;
        a.h[0] = *(const bf16x8*)&Aself[lr][k0 + lh * 8];
        a.h[1] = *(const bf16x8*)&Aself[lr][k0 + lh * 8 + 16];
        const __bf16* bp = &Ws[(size_t)(n0 + lr) * INF + k0 + lh * 8];
        b.h[0] = *(const bf16x8*)bp; b.h[1] = *(const bf16x8*)(bp + 16);
        acc = wmma_bf16(a.v, b.v, acc);
    }
#pragma unroll
    for (int kk = 0; kk < 4; ++kk) {
        const int k0 = kk * 32;
        BF16Frag a, b;
        a.h[0] = *(const bf16x8*)&Anei[lr][k0 + lh * 8];
        a.h[1] = *(const bf16x8*)&Anei[lr][k0 + lh * 8 + 16];
        const __bf16* bp = &Wn[(size_t)(n0 + lr) * INF + k0 + lh * 8];
        b.h[0] = *(const bf16x8*)bp; b.h[1] = *(const bf16x8*)(bp + 16);
        acc = wmma_bf16(a.v, b.v, acc);
    }
    const int col = n0 + lr;
    const float bias = b0[col];
#pragma unroll
    for (int v = 0; v < 8; ++v) {
        int row = v + lh * 8;
        h0[(size_t)(row0 + row) * NH + col] = acc[v] + bias;
    }
}

// ---------------- stage 4: BN stats + apply ----------------
__global__ __launch_bounds__(256) void k_bnstats(const float* __restrict__ h0,
                                                 float* __restrict__ stats) {
    const int c = threadIdx.x; // 256 channels
    float s = 0.f, s2 = 0.f;
    for (int r = blockIdx.x; r < N1v; r += gridDim.x) {
        float v = h0[(size_t)r * NH + c];
        s += v; s2 += v * v;
    }
    atomicAdd(&stats[c], s);
    atomicAdd(&stats[NH + c], s2);
}

__global__ __launch_bounds__(256) void k_bnapply(const float* __restrict__ h0,
                                                 const float* __restrict__ stats,
                                                 const float* __restrict__ gamma,
                                                 const float* __restrict__ beta,
                                                 __bf16* __restrict__ h0n) {
    size_t i = (size_t)blockIdx.x * blockDim.x + threadIdx.x;
    if (i >= (size_t)N1v * NH) return;
    int c = (int)(i & (NH - 1));
    float invN = 1.f / (float)N1v;
    float mu = stats[c] * invN;
    float var = stats[NH + c] * invN - mu * mu;
    float y = (h0[i] - mu) * rsqrtf(var + BN_EPS) * gamma[c] + beta[c];
    y = y > 0.f ? y : 0.f;
    h0n[i] = (__bf16)y;
}

// ---------------- stage 5: LSTM gate GEMM (one time-step) ----------------
// grid (256, 8), block 256 (8 waves). Block computes 16 rows x 128 cols of g (N=1024 total).
// K = 256 (gathered x_t vs W_ih) + 256 (h_prev vs W_hh). Tile staging uses async-to-LDS
// when the toolchain exposes it.
__global__ __launch_bounds__(256) void k_lstm_gemm(const __bf16* __restrict__ h0n,
                                                   const __bf16* __restrict__ hprev,
                                                   const int* __restrict__ nbr1, int t,
                                                   const __bf16* __restrict__ Wih,
                                                   const __bf16* __restrict__ Whh,
                                                   const float* __restrict__ bih,
                                                   const float* __restrict__ bhh,
                                                   float* __restrict__ g) {
    __shared__ __align__(16) __bf16 Ax[16][264];
    __shared__ __align__(16) __bf16 Ah[16][264];
    const int row0 = blockIdx.x * 16;
    const int tid  = threadIdx.x;
    {
        int r = tid >> 4, c0 = (tid & 15) * 16;
        int src = nbr1[(size_t)(row0 + r) * F1v + t];
        const __bf16* gx = &h0n[(size_t)src * NH + c0];
        const __bf16* gh = &hprev[(size_t)(row0 + r) * NH + c0];
#ifdef HAVE_ASYNC_LDS
        async_copy_b128(&Ax[r][c0],     gx);
        async_copy_b128(&Ax[r][c0 + 8], gx + 8);
        async_copy_b128(&Ah[r][c0],     gh);
        async_copy_b128(&Ah[r][c0 + 8], gh + 8);
        wait_async0();
#else
        *(bf16x8*)&Ax[r][c0]     = *(const bf16x8*)gx;
        *(bf16x8*)&Ax[r][c0 + 8] = *(const bf16x8*)(gx + 8);
        *(bf16x8*)&Ah[r][c0]     = *(const bf16x8*)gh;
        *(bf16x8*)&Ah[r][c0 + 8] = *(const bf16x8*)(gh + 8);
#endif
    }
    __syncthreads();
    const int wave = tid >> 5, lane = tid & 31;
    const int n0 = blockIdx.y * 128 + wave * 16;
    const int lr = lane & 15, lh = lane >> 4;
    f32x8 acc = {};
#pragma unroll
    for (int kk = 0; kk < 8; ++kk) {
        const int k0 = kk * 32;
        BF16Frag a, b;
        a.h[0] = *(const bf16x8*)&Ax[lr][k0 + lh * 8];
        a.h[1] = *(const bf16x8*)&Ax[lr][k0 + lh * 8 + 16];
        const __bf16* bp = &Wih[(size_t)(n0 + lr) * NH + k0 + lh * 8];
        b.h[0] = *(const bf16x8*)bp; b.h[1] = *(const bf16x8*)(bp + 16);
        acc = wmma_bf16(a.v, b.v, acc);
    }
#pragma unroll
    for (int kk = 0; kk < 8; ++kk) {
        const int k0 = kk * 32;
        BF16Frag a, b;
        a.h[0] = *(const bf16x8*)&Ah[lr][k0 + lh * 8];
        a.h[1] = *(const bf16x8*)&Ah[lr][k0 + lh * 8 + 16];
        const __bf16* bp = &Whh[(size_t)(n0 + lr) * NH + k0 + lh * 8];
        b.h[0] = *(const bf16x8*)bp; b.h[1] = *(const bf16x8*)(bp + 16);
        acc = wmma_bf16(a.v, b.v, acc);
    }
    const int col = n0 + lr;
    const float bias = bih[col] + bhh[col];
#pragma unroll
    for (int v = 0; v < 8; ++v) {
        int row = v + lh * 8;
        g[(size_t)(row0 + row) * NG + col] = acc[v] + bias;
    }
}

// ---------------- stage 6: LSTM cell elementwise ----------------
__global__ __launch_bounds__(256) void k_lstm_cell(const float* __restrict__ g,
                                                   float* __restrict__ c,
                                                   __bf16* __restrict__ h) {
    size_t i = (size_t)blockIdx.x * blockDim.x + threadIdx.x;
    if (i >= (size_t)Bv * NH) return;
    int row = (int)(i >> 8), j = (int)(i & 255);
    const float* gr = &g[(size_t)row * NG];
    float ig = sigm(gr[j]);
    float fg = sigm(gr[NH + j]);
    float gg = tanhf(gr[2 * NH + j]);
    float og = sigm(gr[3 * NH + j]);
    float cn = fg * c[i] + ig * gg;
    c[i] = cn;
    h[i] = (__bf16)(og * tanhf(cn));
}

// ---------------- stage 7: out = h0n[:B] @ Ws1^T + hT @ Wn1^T + b1 (fp32 out) ------------
// block 64 thr (2 waves), 16 rows x 32 cols, K = 256 + 256.
__global__ __launch_bounds__(64) void k_final(const __bf16* __restrict__ h0n,
                                              const __bf16* __restrict__ hT,
                                              const __bf16* __restrict__ Ws1,
                                              const __bf16* __restrict__ Wn1,
                                              const float* __restrict__ b1,
                                              float* __restrict__ out) {
    __shared__ __align__(16) __bf16 Aself[16][264];
    __shared__ __align__(16) __bf16 Ah[16][264];
    const int row0 = blockIdx.x * 16;
    const int tid  = threadIdx.x;
    for (int e = tid; e < 256; e += 64) {
        int r = e >> 4, c0 = (e & 15) * 16;
        *(bf16x8*)&Aself[r][c0]     = *(const bf16x8*)&h0n[(size_t)(row0 + r) * NH + c0];
        *(bf16x8*)&Aself[r][c0 + 8] = *(const bf16x8*)&h0n[(size_t)(row0 + r) * NH + c0 + 8];
        *(bf16x8*)&Ah[r][c0]        = *(const bf16x8*)&hT[(size_t)(row0 + r) * NH + c0];
        *(bf16x8*)&Ah[r][c0 + 8]    = *(const bf16x8*)&hT[(size_t)(row0 + r) * NH + c0 + 8];
    }
    __syncthreads();
    const int wave = tid >> 5, lane = tid & 31;
    const int n0 = wave * 16;
    const int lr = lane & 15, lh = lane >> 4;
    f32x8 acc = {};
#pragma unroll
    for (int kk = 0; kk < 8; ++kk) {
        const int k0 = kk * 32;
        BF16Frag a, b;
        a.h[0] = *(const bf16x8*)&Aself[lr][k0 + lh * 8];
        a.h[1] = *(const bf16x8*)&Aself[lr][k0 + lh * 8 + 16];
        const __bf16* bp = &Ws1[(size_t)(n0 + lr) * NH + k0 + lh * 8];
        b.h[0] = *(const bf16x8*)bp; b.h[1] = *(const bf16x8*)(bp + 16);
        acc = wmma_bf16(a.v, b.v, acc);
    }
#pragma unroll
    for (int kk = 0; kk < 8; ++kk) {
        const int k0 = kk * 32;
        BF16Frag a, b;
        a.h[0] = *(const bf16x8*)&Ah[lr][k0 + lh * 8];
        a.h[1] = *(const bf16x8*)&Ah[lr][k0 + lh * 8 + 16];
        const __bf16* bp = &Wn1[(size_t)(n0 + lr) * NH + k0 + lh * 8];
        b.h[0] = *(const bf16x8*)bp; b.h[1] = *(const bf16x8*)(bp + 16);
        acc = wmma_bf16(a.v, b.v, acc);
    }
    const int col = n0 + lr;
    const float bias = b1[col];
#pragma unroll
    for (int v = 0; v < 8; ++v) {
        int row = v + lh * 8;
        out[(size_t)(row0 + row) * NC + col] = acc[v] + bias;
    }
}

// ---------------- launcher ----------------
extern "C" void kernel_launch(void* const* d_in, const int* in_sizes, int n_in,
                              void* d_out, int out_size, void* d_ws, size_t ws_size,
                              hipStream_t stream) {
    (void)in_sizes; (void)n_in; (void)out_size; (void)ws_size;
    const float* x       = (const float*)d_in[0];
    const int*   nbr0    = (const int*)d_in[1];
    const int*   nbr1    = (const int*)d_in[2];
    const float* W_pool  = (const float*)d_in[3];
    const float* b_pool  = (const float*)d_in[4];
    const float* W_self0 = (const float*)d_in[5];
    const float* W_neigh0= (const float*)d_in[6];
    const float* b0      = (const float*)d_in[7];
    const float* gamma0  = (const float*)d_in[8];
    const float* beta0   = (const float*)d_in[9];
    const float* W_ih    = (const float*)d_in[10];
    const float* W_hh    = (const float*)d_in[11];
    const float* b_ih    = (const float*)d_in[12];
    const float* b_hh    = (const float*)d_in[13];
    const float* W_self1 = (const float*)d_in[14];
    const float* W_neigh1= (const float*)d_in[15];
    const float* b1      = (const float*)d_in[16];
    float* out = (float*)d_out;

    char* ws = (char*)d_ws;
    size_t off = 0;
    auto alloc = [&](size_t bytes) -> void* {
        void* p = ws + off;
        off += (bytes + 255) & ~(size_t)255;
        return p;
    };

    __bf16* hp      = (__bf16*)alloc((size_t)N0v * INF * 2);   // pooled MLP out
    __bf16* hneigh  = (__bf16*)alloc((size_t)N1v * INF * 2);
    float*  h0      = (float*) alloc((size_t)N1v * NH * 4);
    __bf16* h0n     = (__bf16*)alloc((size_t)N1v * NH * 2);
    float*  gbuf    = (float*) alloc((size_t)Bv * NG * 4);
    // zero region (contiguous): LSTM h (bf16), c (f32), BN stats
    __bf16* hlstm   = (__bf16*)alloc((size_t)Bv * NH * 2);
    float*  cbuf    = (float*) alloc((size_t)Bv * NH * 4);
    float*  stats   = (float*) alloc(2 * NH * 4);
    // bf16 weights
    __bf16* Wp_b  = (__bf16*)alloc((size_t)INF * INF * 2);
    __bf16* Ws0_b = (__bf16*)alloc((size_t)NH * INF * 2);
    __bf16* Wn0_b = (__bf16*)alloc((size_t)NH * INF * 2);
    __bf16* Wih_b = (__bf16*)alloc((size_t)NG * NH * 2);
    __bf16* Whh_b = (__bf16*)alloc((size_t)NG * NH * 2);
    __bf16* Ws1_b = (__bf16*)alloc((size_t)NC * NH * 2);
    __bf16* Wn1_b = (__bf16*)alloc((size_t)NC * NH * 2);

    // weight conversion
    k_cvt_bf16<<<(INF * INF + 255) / 256, 256, 0, stream>>>(W_pool, Wp_b, INF * INF);
    k_cvt_bf16<<<(NH * INF + 255) / 256, 256, 0, stream>>>(W_self0, Ws0_b, NH * INF);
    k_cvt_bf16<<<(NH * INF + 255) / 256, 256, 0, stream>>>(W_neigh0, Wn0_b, NH * INF);
    k_cvt_bf16<<<(NG * NH + 255) / 256, 256, 0, stream>>>(W_ih, Wih_b, NG * NH);
    k_cvt_bf16<<<(NG * NH + 255) / 256, 256, 0, stream>>>(W_hh, Whh_b, NG * NH);
    k_cvt_bf16<<<(NC * NH + 255) / 256, 256, 0, stream>>>(W_self1, Ws1_b, NC * NH);
    k_cvt_bf16<<<(NC * NH + 255) / 256, 256, 0, stream>>>(W_neigh1, Wn1_b, NC * NH);

    // zero LSTM h/c and BN stats (contiguous region starting at hlstm)
    {
        size_t zbytes = (size_t)Bv * NH * 2 + (size_t)Bv * NH * 4 + 2 * NH * 4;
        size_t n16 = (zbytes + 15) / 16;
        k_zero16<<<1024, 256, 0, stream>>>((uint4*)hlstm, n16);
    }

    // stage 1: pooled MLP over all N0 rows
    k_pool<<<N0v / 16, 256, 0, stream>>>(x, Wp_b, b_pool, hp);
    // stage 2: neighbor max
    k_max<<<N1v, 128, 0, stream>>>(hp, nbr0, hneigh);
    // stage 3: layer0 combine
    k_layer0<<<N1v / 16, 512, 0, stream>>>(x, hneigh, Ws0_b, Wn0_b, b0, h0);
    // stage 4: BN
    k_bnstats<<<256, 256, 0, stream>>>(h0, stats);
    k_bnapply<<<(N1v * NH + 255) / 256, 256, 0, stream>>>(h0, stats, gamma0, beta0, h0n);
    // stage 5/6: LSTM over 10 neighbor steps
    for (int t = 0; t < F1v; ++t) {
        k_lstm_gemm<<<dim3(Bv / 16, NG / 128), 256, 0, stream>>>(
            h0n, hlstm, nbr1, t, Wih_b, Whh_b, b_ih, b_hh, gbuf);
        k_lstm_cell<<<(Bv * NH + 255) / 256, 256, 0, stream>>>(gbuf, cbuf, hlstm);
    }
    // stage 7: final projection
    k_final<<<Bv / 16, 64, 0, stream>>>(h0n, hlstm, Ws1_b, Wn1_b, b1, out);
}
